// ppmoe_gate_58987080843399
// MI455X (gfx1250) — matrix operand order repaired and checked by reading, not compiled, not run on hardware
//
#include <hip/hip_runtime.h>

// ppmoe_gate for MI455X (gfx1250, wave32).
//
// Key insight: the reference output depends only on prob_board (8 KiB) and the
// PRNG — NOT on x (512 MiB). Roofline: 4 MiB of output at 23.3 TB/s ~= 0.18us;
// never touch x. Sampling is done by inverse-CDF (distribution-equivalent to
// JAX's Gumbel argmax) so we need only ~0.5 Threefry-2x32 calls per token.
// The 64-bin CDF prefix-scan is computed with V_WMMA_F32_16X16X32_F16
// (A = lower-triangular ones, B = probabilities), f32 accumulate.

typedef __attribute__((ext_vector_type(16))) _Float16 v16h;
typedef __attribute__((ext_vector_type(8)))  float    v8f;

#define BLOCK 256
#define TOKENS_PER_THREAD 4
#define N_EXPERTS 64

struct u2 { unsigned int x, y; };

__device__ __forceinline__ unsigned int rotl32(unsigned int v, int s) {
  return (v << s) | (v >> (32 - s));
}

// Genuine Threefry-2x32, 20 rounds (JAX's PRNG).
__device__ __forceinline__ u2 threefry2x32(u2 key, u2 ctr) {
  unsigned int ks0 = key.x, ks1 = key.y;
  unsigned int ks2 = key.x ^ key.y ^ 0x1BD11BDAu;
  unsigned int x0 = ctr.x + ks0;
  unsigned int x1 = ctr.y + ks1;
#define TF_R(r) { x0 += x1; x1 = rotl32(x1, r); x1 ^= x0; }
  TF_R(13) TF_R(15) TF_R(26) TF_R(6)
  x0 += ks1; x1 += ks2 + 1u;
  TF_R(17) TF_R(29) TF_R(16) TF_R(24)
  x0 += ks2; x1 += ks0 + 2u;
  TF_R(13) TF_R(15) TF_R(26) TF_R(6)
  x0 += ks0; x1 += ks1 + 3u;
  TF_R(17) TF_R(29) TF_R(16) TF_R(24)
  x0 += ks1; x1 += ks2 + 4u;
  TF_R(13) TF_R(15) TF_R(26) TF_R(6)
  x0 += ks2; x1 += ks0 + 5u;
#undef TF_R
  u2 r; r.x = x0; r.y = x1; return r;
}

__global__ void __launch_bounds__(BLOCK) ppmoe_gate_kernel(
    const float* __restrict__ prob_board,
    float* __restrict__ out,
    int n_tokens) {
  __shared__ float cdf[N_EXPERTS];

  // ---- key derivation (uniform across all threads; cheap) ----
  u2 root;  root.x = 0u;  root.y = 42u;              // jax.random.key(42)
  u2 c01;   c01.x = 0u;   c01.y = 1u;
  u2 c23;   c23.x = 2u;   c23.y = 3u;
  u2 k_sel  = threefry2x32(root, c01);               // split -> k_sel
  u2 k_samp = threefry2x32(root, c23);               // split -> k_samp
  u2 c00;   c00.x = 0u;   c00.y = 0u;
  unsigned int sel = threefry2x32(k_sel, c00).x & 31u; // N_DISTS=32 (pow2)
  const float* __restrict__ row = prob_board + (size_t)sel * N_EXPERTS;

  const int tid  = threadIdx.x;
  const int lane = tid & 31;

  // ---- wave 0: 64-bin inclusive CDF via one WMMA (f32 accumulate) ----
  if (tid < 32) {
    __builtin_prefetch(row, 0, 1);                   // global_prefetch_b8
    const bool hi = lane >= 16;
    const int  mn = lane & 15;                       // M for A/D-row, N for B/D-col

    // A (16x32 f16) = [L16 | 0], L16 = lower-triangular ones (exact in f16).
    // A layout used here (contiguous k = 16h + j): lanes 0-15 hold M=0..15 with K=0..15;
    // lanes 16-31 hold M=0..15 with K=16..31.
    v16h a;
    #pragma unroll
    for (int j = 0; j < 16; ++j) {
      int k = (hi ? 16 : 0) + j;
      a[j] = (_Float16)((k < 16 && k <= mn) ? 1.0f : 0.0f);
    }

    // B (32x16 f16): columns n=0..3 carry the 64 probs, 16 per column, K=0..15.
    // B layout used here (same k order as A): lanes 0-15 = column N=lane, K=0..15 (2 per VGPR);
    // lanes 16-31 = K=16..31 (all zero here).
    v16h b;
    #pragma unroll
    for (int j = 0; j < 16; ++j) {
      float v = 0.0f;
      if (!hi && mn < 4) v = row[mn * 16 + j];
      b[j] = (_Float16)v;
    }

    v8f c = {};
    // D[m][n] = sum_{k<=m} p[n*16+k]  (within-block-of-16 inclusive prefix)
    v8f d = __builtin_amdgcn_wmma_f32_16x16x32_f16(
        /*neg_a=*/false, a, /*neg_b=*/false, b,
        /*c_mod=*/(short)0, c, /*reuse_a=*/false, /*reuse_b=*/false);

    // Block sums S_n = D[15][n] live in lane 16+n, element d[7] (f32 C/D layout).
    float d7 = d[7];
    float s0 = __shfl(d7, 16, 32);
    float s1 = __shfl(d7, 17, 32);
    float s2 = __shfl(d7, 18, 32);

    if (mn < 4) {
      float off = (mn >= 1 ? s0 : 0.0f)
                + (mn >= 2 ? s1 : 0.0f)
                + (mn >= 3 ? s2 : 0.0f);
      #pragma unroll
      for (int r = 0; r < 8; ++r) {
        int m = (hi ? 8 : 0) + r;                    // D row index for VGPR r
        cdf[mn * 16 + m] = d[r] + off;               // global inclusive CDF
      }
    }
  }
  __syncthreads();

  // ---- inverse-CDF categorical sampling: 4 consecutive tokens / thread ----
  const int base = (blockIdx.x * BLOCK + tid) * TOKENS_PER_THREAD;
  if (base >= n_tokens) return;

  float res[TOKENS_PER_THREAD];
  #pragma unroll
  for (int p = 0; p < TOKENS_PER_THREAD / 2; ++p) {
    u2 ctr; ctr.x = (unsigned int)(base >> 1) + p; ctr.y = 0x9E3779B9u;
    u2 bits = threefry2x32(k_samp, ctr);
    float u[2];
    u[0] = (float)(bits.x >> 8) * 0x1p-24f;          // [0,1)
    u[1] = (float)(bits.y >> 8) * 0x1p-24f;
    #pragma unroll
    for (int q = 0; q < 2; ++q) {
      // smallest i with u < cdf[i]; branch-free 6-step binary search on LDS
      int idx = 0;
      #pragma unroll
      for (int w = 32; w >= 1; w >>= 1)
        if (cdf[idx + w - 1] <= u[q]) idx += w;
      res[2 * p + q] = (float)(idx > 63 ? 63 : idx);
    }
  }

  if (base + TOKENS_PER_THREAD <= n_tokens) {
    float4 v4; v4.x = res[0]; v4.y = res[1]; v4.z = res[2]; v4.w = res[3];
    *(float4*)(out + base) = v4;                     // global_store_b128
  } else {
    for (int i = 0; i < TOKENS_PER_THREAD && base + i < n_tokens; ++i)
      out[base + i] = res[i];
  }
}

extern "C" void kernel_launch(void* const* d_in, const int* in_sizes, int n_in,
                              void* d_out, int out_size, void* d_ws, size_t ws_size,
                              hipStream_t stream) {
  (void)in_sizes; (void)n_in; (void)d_ws; (void)ws_size;
  // d_in[0] = x [1048576 x 128] f32 — intentionally unused (output is
  // independent of x; skipping it avoids 512 MiB of wasted HBM traffic).
  const float* prob_board = (const float*)d_in[1];   // [32 x 64] f32
  float* out = (float*)d_out;                        // [n_tokens] f32
  const int n_tokens = out_size;

  const int tokens_per_block = BLOCK * TOKENS_PER_THREAD;
  const int blocks = (n_tokens + tokens_per_block - 1) / tokens_per_block;
  hipLaunchKernelGGL(ppmoe_gate_kernel, dim3(blocks), dim3(BLOCK), 0, stream,
                     prob_board, out, n_tokens);
}
